// CrossAttention_17910013624927
// MI455X (gfx1250) — matrix operand order repaired and checked by reading, not compiled, not run on hardware
//
#include <hip/hip_runtime.h>

typedef _Float16 v16h __attribute__((ext_vector_type(16)));
typedef v16h v16h_u __attribute__((aligned(16)));   // LDS rows are only 16B aligned
typedef float v8f __attribute__((ext_vector_type(8)));

namespace {
constexpr int kB   = 2;
constexpr int kTq  = 2048;
constexpr int kTk  = 2048;
constexpr int kH   = 32;
constexpr int kHkv = 8;
constexpr int kD   = 128;
constexpr int kGq  = kH / kHkv;     // 4 q-heads per kv-head
constexpr int QT   = 64;            // q rows per block (16 per wave)
constexpr int KT   = 32;            // keys per iteration
constexpr float kNeg   = -10000.0f;
constexpr float kScale = 0.08838834764831845f;  // 1/sqrt(128)

constexpr int KSTR = kD + 8;   // kbuf row stride (halves): 272B rows, banks shift 4/row
constexpr int VSTR = KT + 8;   // vbufT row stride (halves): 80B rows, banks shift 20/row
constexpr int PSTR = KT + 8;   // pbuf row stride (halves)
}

__global__ __launch_bounds__(128)
void fattn_rc_kernel(const float* __restrict__ q,
                     const float* __restrict__ kv,
                     const unsigned char* __restrict__ kpm,
                     float* __restrict__ out) {
  __shared__ _Float16 kbuf[KT * KSTR];        // [key][d]        (f16)
  __shared__ _Float16 vbufT[kD * VSTR];       // [d][key]        (f16, transposed)
  __shared__ _Float16 pbuf[4 * 16 * PSTR];    // per-wave P tile [row][key]

  const int q0  = blockIdx.x * QT;
  const int h   = blockIdx.y;
  const int b   = blockIdx.z;
  const int hkv = h / kGq;                    // jnp.repeat(axis=3): h -> h/4

  const int tid  = threadIdx.x;
  const int wave = tid >> 5;
  const int lane = tid & 31;
  const int hlf  = lane >> 4;                 // which 16-lane half
  const int col  = lane & 15;

  // ---- Q tile for this wave into WMMA A-layout fragments (pre-scaled) ----
  const int qrow0 = q0 + wave * 16;
  const int myrow = qrow0 + col;              // A layout: row = lane % 16
  const float* qp = q + (((size_t)b * kTq + myrow) * kH + h) * kD;
  v16h aq[4];
#pragma unroll
  for (int c = 0; c < 4; ++c) {
#pragma unroll
    for (int e = 0; e < 16; e += 2) {
      const int k = e + (hlf << 4);  // contiguous A-frag K map: k = e + 16*hlf
      const int d = c * 32 + k;
      aq[c][e]     = (_Float16)(qp[d]     * kScale);
      aq[c][e + 1] = (_Float16)(qp[d + 1] * kScale);
    }
  }

  v8f  o[8];
  float m[8], l[8];
#pragma unroll
  for (int f = 0; f < 8; ++f)
#pragma unroll
    for (int r = 0; r < 8; ++r) o[f][r] = 0.0f;
#pragma unroll
  for (int r = 0; r < 8; ++r) { m[r] = -3.0e38f; l[r] = 0.0f; }

  // Reference keeps scores where col > row (reversed-causal). Smallest useful
  // key for this tile is q0+1. The last q tile contains the fully-masked row
  // (qg == Tk-1) whose softmax is uniform over ALL keys -> scan everything.
  const int kt0 = (q0 + QT >= kTk) ? 0 : (q0 >> 5);

  _Float16* pwave = &pbuf[wave * 16 * PSTR];

  for (int kt = kt0; kt < kTk / KT; ++kt) {
    const int kb = kt * KT;
    __syncthreads();                          // previous iteration's readers done

    // ---- cooperative stage: K row-major, V transposed, f32 -> f16 ----
#pragma unroll
    for (int i = 0; i < 8; ++i) {
      const int idx = tid + i * 128;
      const int row = idx >> 5;               // key within tile, 0..31
      const int d4  = (idx & 31) << 2;        // d offset, 0..124
      const float* kp = kv + ((((size_t)b * kTk + kb + row) * 2 + 0) * kHkv + hkv) * kD + d4;
      const float4 kk = *(const float4*)kp;
      const float4 vv = *(const float4*)(kp + kHkv * kD);   // V plane
      _Float16* kd = &kbuf[row * KSTR + d4];
      kd[0] = (_Float16)kk.x; kd[1] = (_Float16)kk.y;
      kd[2] = (_Float16)kk.z; kd[3] = (_Float16)kk.w;
      vbufT[(d4 + 0) * VSTR + row] = (_Float16)vv.x;
      vbufT[(d4 + 1) * VSTR + row] = (_Float16)vv.y;
      vbufT[(d4 + 2) * VSTR + row] = (_Float16)vv.z;
      vbufT[(d4 + 3) * VSTR + row] = (_Float16)vv.w;
    }
    __syncthreads();

    // ---- S = Q * K^T : two 16x16 f32 tiles (keys 0-15 / 16-31) ----
    v8f s0, s1;
#pragma unroll
    for (int r = 0; r < 8; ++r) { s0[r] = 0.0f; s1[r] = 0.0f; }
#pragma unroll
    for (int c = 0; c < 4; ++c) {
      // B-frag: N = key = col, K(e) = d = c*32 + e + 16*hlf  (contiguous in kbuf row)
      const v16h bk0 = *(const v16h_u*)&kbuf[col        * KSTR + c * 32 + (hlf << 4)];
      const v16h bk1 = *(const v16h_u*)&kbuf[(col + 16) * KSTR + c * 32 + (hlf << 4)];
      s0 = __builtin_amdgcn_wmma_f32_16x16x32_f16(false, aq[c], false, bk0, (short)0, s0, false, false);
      s1 = __builtin_amdgcn_wmma_f32_16x16x32_f16(false, aq[c], false, bk1, (short)0, s1, false, false);
    }

    // ---- mask + online softmax; P -> LDS (f16) ----
    const int kg0 = kb + col, kg1 = kb + 16 + col;
    const float pad0 = kpm[(size_t)b * kTk + kg0] ? 0.0f : kNeg;
    const float pad1 = kpm[(size_t)b * kTk + kg1] ? 0.0f : kNeg;
#pragma unroll
    for (int r = 0; r < 8; ++r) {
      const int qg = qrow0 + (hlf << 3) + r;  // C layout: row = r + 8*hlf
      float x0 = (kg0 > qg) ? (s0[r] + pad0) : kNeg;
      float x1 = (kg1 > qg) ? (s1[r] + pad1) : kNeg;
      float mx = fmaxf(x0, x1);
      mx = fmaxf(mx, __shfl_xor(mx, 1));
      mx = fmaxf(mx, __shfl_xor(mx, 2));
      mx = fmaxf(mx, __shfl_xor(mx, 4));
      mx = fmaxf(mx, __shfl_xor(mx, 8));      // within 16-lane half
      const float mn = fmaxf(m[r], mx);
      const float al = __expf(m[r] - mn);
      const float p0 = __expf(x0 - mn);
      const float p1 = __expf(x1 - mn);
      float rs = p0 + p1;
      rs += __shfl_xor(rs, 1);
      rs += __shfl_xor(rs, 2);
      rs += __shfl_xor(rs, 4);
      rs += __shfl_xor(rs, 8);
      l[r] = l[r] * al + rs;
      m[r] = mn;
#pragma unroll
      for (int f = 0; f < 8; ++f) o[f][r] *= al;
      _Float16* pr = &pwave[((hlf << 3) + r) * PSTR];
      pr[col]      = (_Float16)p0;
      pr[col + 16] = (_Float16)p1;
    }
    __syncthreads();                          // uniform across the block

    // ---- O += P * V ----
    v16h ap;
    const _Float16* prd = &pwave[col * PSTR]; // A layout: row = lane % 16
#pragma unroll
    for (int e = 0; e < 16; e += 2) {
      const int k = e + (hlf << 4);
      ap[e]     = prd[k];
      ap[e + 1] = prd[k + 1];
    }
#pragma unroll
    for (int f = 0; f < 8; ++f) {
      // B-frag: N = d-col = f*16+col, K(e) = key = e + 16*hlf (contiguous in vbufT row)
      const v16h bv = *(const v16h_u*)&vbufT[(f * 16 + col) * VSTR + (hlf << 4)];
      o[f] = __builtin_amdgcn_wmma_f32_16x16x32_f16(false, ap, false, bv, (short)0, o[f], false, false);
    }
  }

  // ---- normalize and store (fp32 out, shape B,Tq,H,D) ----
#pragma unroll
  for (int r = 0; r < 8; ++r) {
    const int qg = qrow0 + (hlf << 3) + r;
    const float inv = 1.0f / l[r];
    float* op = out + (((size_t)b * kTq + qg) * kH + h) * kD + col;
#pragma unroll
    for (int f = 0; f < 8; ++f) op[f * 16] = o[f][r] * inv;
  }
}

extern "C" void kernel_launch(void* const* d_in, const int* in_sizes, int n_in,
                              void* d_out, int out_size, void* d_ws, size_t ws_size,
                              hipStream_t stream) {
  (void)in_sizes; (void)n_in; (void)out_size; (void)d_ws; (void)ws_size;
  const float* q  = (const float*)d_in[0];
  const float* kv = (const float*)d_in[1];
  const unsigned char* kpm = (const unsigned char*)d_in[2];
  float* out = (float*)d_out;
  dim3 grid(kTq / QT, kH, kB);
  fattn_rc_kernel<<<grid, dim3(128), 0, stream>>>(q, kv, kpm, out);
}
